// UnimolFeatureExtractor_1503238553650
// MI455X (gfx1250) — matrix operand order repaired and checked by reading, not compiled, not run on hardware
//
#include <hip/hip_runtime.h>

// ---------------------------------------------------------------------------
// MI455X (gfx1250, wave32) implementation of the 5-stage beam-search MLP.
// Dominant cost: bf16 WMMA GEMMs (v_wmma_f32_16x16x32_bf16), fp32 accumulate.
// GEMM is register-staged + LDS double-buffered so next tile's global loads
// are in flight while the current tile's 8 WMMAs execute. B-tile rows are
// clamped (not predicated): out-of-range rows feed only discarded columns.
// ---------------------------------------------------------------------------

typedef unsigned short ushort_t;
typedef __attribute__((ext_vector_type(16))) __bf16 v16bf;
typedef __attribute__((ext_vector_type(8)))  float  v8f;

struct alignas(16) U4 { unsigned int x, y, z, w; };

__device__ __forceinline__ ushort_t f2bf(float f) {
    unsigned int u = __float_as_uint(f);
    unsigned int r = u + 0x7FFFu + ((u >> 16) & 1u);   // round-to-nearest-even
    return (ushort_t)(r >> 16);
}

// ------------------------- conversion kernels ------------------------------

__global__ void k_cvt_bf16(const float* __restrict__ in, ushort_t* __restrict__ out, int n) {
    int i = blockIdx.x * blockDim.x + threadIdx.x;
    if (i < n) out[i] = f2bf(in[i]);
}

// W: [K,N] fp32 row-major  ->  Wt: [N,K] bf16 (so B tiles are K-contiguous)
__global__ void k_transpose_bf16(const float* __restrict__ W, ushort_t* __restrict__ Wt,
                                 int K, int N) {
    int i = blockIdx.x * blockDim.x + threadIdx.x;
    if (i >= N * K) return;
    int n = i / K, k = i - n * K;
    Wt[(size_t)n * K + k] = f2bf(W[(size_t)k * N + n]);
}

// ------------------------------- WMMA GEMM ---------------------------------
// C[M,N] = act(A[M,K](bf16) * Bt[N,K](bf16)^T + bias)
// ACT: 0 = relu -> bf16 out, 1 = tanh -> bf16 out, 2 = none -> fp32 out
// Block tile 128x128, K-step 32; 8 waves as 4(M) x 2(N), wave tile 32x64.

#define BM 128
#define BN 128
#define BK 32
#define LDSS 48                    // halfword row stride: 96 B -> 16B-aligned b128
#define BUFH ((BM + BN) * LDSS)    // halfwords per double-buffer slot

union BFrag { U4 u[2]; v16bf v; };

template <int ACT>
__global__ __launch_bounds__(256) void k_gemm_bf16(
    const ushort_t* __restrict__ A, const ushort_t* __restrict__ Bt,
    const float* __restrict__ bias, void* __restrict__ Cv,
    int M, int N, int K)
{
    __shared__ __align__(16) ushort_t smem[2 * BUFH];

    const int tid  = threadIdx.x;
    const int lane = tid & 31;
    const int wid  = tid >> 5;
    const int wm   = wid & 3;         // 4 waves along M (32 rows each)
    const int wn   = wid >> 2;        // 2 waves along N (64 cols each)
    const int m0   = blockIdx.y * BM;
    const int n0   = blockIdx.x * BN;

    // ---- per-thread tile-load assignment: chunks tid and tid+256 of 512
    const int rowA0 = tid >> 2,        ccA0 = tid & 3;
    const int rowA1 = rowA0 + 64;
    const int offA0 = rowA0 * LDSS + ccA0 * 8;
    const int offA1 = rowA1 * LDSS + ccA0 * 8;
    const int offB0 = BM * LDSS + offA0;
    const int offB1 = BM * LDSS + offA1;
    // clamp B rows: rows >= N only feed output columns >= N (never stored)
    const int rB0 = min(n0 + rowA0, N - 1);
    const int rB1 = min(n0 + rowA1, N - 1);

    const ushort_t* gA0 = A  + (size_t)(m0 + rowA0) * K + ccA0 * 8;
    const ushort_t* gA1 = A  + (size_t)(m0 + rowA1) * K + ccA0 * 8;
    const ushort_t* gB0 = Bt + (size_t)rB0 * K + ccA0 * 8;
    const ushort_t* gB1 = Bt + (size_t)rB1 * K + ccA0 * 8;

    // ---- fragment LDS offsets (ISA 7.12.2 16-bit layouts), hoisted
    const int klo = (lane < 16) ? 0 : 8;      // A: K {0..7,16..23} / {8..15,24..31}
    const int kb  = (lane < 16) ? 0 : 8;      // B: K {0..7,16..23} / {8..15,24..31}
    int foA[2], foB[4];
    #pragma unroll
    for (int i = 0; i < 2; i++)
        foA[i] = (wm * 32 + i * 16 + (lane & 15)) * LDSS + klo;
    #pragma unroll
    for (int j = 0; j < 4; j++)
        foB[j] = BM * LDSS + (wn * 64 + j * 16 + (lane & 15)) * LDSS + kb;

    v8f acc[2][4];
    #pragma unroll
    for (int i = 0; i < 2; i++)
        #pragma unroll
        for (int j = 0; j < 4; j++)
            #pragma unroll
            for (int v = 0; v < 8; v++) acc[i][j][v] = 0.0f;

    // ---- prime the register stage with the first K-tile
    U4 ra0 = *reinterpret_cast<const U4*>(gA0);
    U4 ra1 = *reinterpret_cast<const U4*>(gA1);
    U4 rb0 = *reinterpret_cast<const U4*>(gB0);
    U4 rb1 = *reinterpret_cast<const U4*>(gB1);

    int p = 0;
    for (int kt = 0; kt < K; kt += BK) {
        const int base = p * BUFH;
        // stage -> LDS buffer p
        *reinterpret_cast<U4*>(&smem[base + offA0]) = ra0;
        *reinterpret_cast<U4*>(&smem[base + offA1]) = ra1;
        *reinterpret_cast<U4*>(&smem[base + offB0]) = rb0;
        *reinterpret_cast<U4*>(&smem[base + offB1]) = rb1;
        __syncthreads();

        // issue next tile's global loads; they stay in flight across the WMMAs
        if (kt + BK < K) {
            gA0 += BK; gA1 += BK; gB0 += BK; gB1 += BK;
            ra0 = *reinterpret_cast<const U4*>(gA0);
            ra1 = *reinterpret_cast<const U4*>(gA1);
            rb0 = *reinterpret_cast<const U4*>(gB0);
            rb1 = *reinterpret_cast<const U4*>(gB1);
        }

        // fragments + 8 WMMAs from buffer p
        BFrag fa[2], fb[4];
        #pragma unroll
        for (int i = 0; i < 2; i++) {
            const U4* q = reinterpret_cast<const U4*>(&smem[base + foA[i]]);
            fa[i].u[0] = q[0];      // +0 halfwords
            fa[i].u[1] = q[2];      // +16 halfwords
        }
        #pragma unroll
        for (int j = 0; j < 4; j++) {
            const U4* q = reinterpret_cast<const U4*>(&smem[base + foB[j]]);
            fb[j].u[0] = q[0];
            fb[j].u[1] = q[2];
        }
        #pragma unroll
        for (int i = 0; i < 2; i++)
            #pragma unroll
            for (int j = 0; j < 4; j++)
                acc[i][j] = __builtin_amdgcn_wmma_f32_16x16x32_bf16(
                    false, fa[i].v, false, fb[j].v, (short)0, acc[i][j], false, false);
        p ^= 1;
    }

    // ---- epilogue: bias + activation; C/D layout: VGPR v -> M = v + 8*(lane>=16)
    const int rbase0 = m0 + wm * 32 + ((lane >> 4) << 3);
    const int cb     = n0 + wn * 64 + (lane & 15);
    #pragma unroll
    for (int i = 0; i < 2; i++) {
        int rbase = rbase0 + i * 16;
        #pragma unroll
        for (int j = 0; j < 4; j++) {
            int col = cb + j * 16;
            if (col >= N) continue;
            float bb = bias[col];
            #pragma unroll
            for (int v = 0; v < 8; v++) {
                float val = acc[i][j][v] + bb;
                size_t idx = (size_t)(rbase + v) * N + col;
                if (ACT == 0)      ((ushort_t*)Cv)[idx] = f2bf(fmaxf(val, 0.0f));
                else if (ACT == 1) ((ushort_t*)Cv)[idx] = f2bf(tanhf(val));
                else               ((float*)Cv)[idx] = val;
            }
        }
    }
}

// --------------------- stage input assembly (gathers) ----------------------
// X row r (= b*n + j) = [ feat_bf16(1024) | relu(W_s[idx_s]+b_s) (256) ... ]

__global__ __launch_bounds__(256) void k_assemble(
    ushort_t* __restrict__ X, const ushort_t* __restrict__ featbf,
    int K, int n, int nmsg,
    const int* i0p, int d0, int w0, const float* W0, const float* B0,
    const int* i1p, int d1, int w1, const float* W1, const float* B1,
    const int* i2p, int d2, int w2, const float* W2, const float* B2,
    const int* i3p, int d3, int w3, const float* W3, const float* B3)
{
    int r = blockIdx.x;
    int t = threadIdx.x;
    int b = r / n, j = r - b * n;
    ushort_t* xr = X + (size_t)r * K;
    const ushort_t* fr = featbf + (size_t)b * 1024;
    for (int q = t; q < 1024; q += 256) xr[q] = fr[q];

    const int*   ip[4] = {i0p, i1p, i2p, i3p};
    const int    dv[4] = {d0, d1, d2, d3};
    const int    wv[4] = {w0, w1, w2, w3};
    const float* Wv[4] = {W0, W1, W2, W3};
    const float* Bv[4] = {B0, B1, B2, B3};
    for (int s = 0; s < nmsg; s++) {
        int idx = ip[s][b * wv[s] + (j / dv[s])];
        float val = Wv[s][(size_t)idx * 256 + t] + Bv[s][t];
        xr[1024 + s * 256 + t] = f2bf(fmaxf(val, 0.0f));
    }
}

// ----------------------- softmax + top-k (wave/row) ------------------------

__global__ __launch_bounds__(256) void k_softmax_topk(
    const float* __restrict__ logits, int R, int D, int k,
    float* __restrict__ sc, int* __restrict__ ix)
{
    int wid  = threadIdx.x >> 5;
    int lane = threadIdx.x & 31;
    int r = blockIdx.x * 8 + wid;
    if (r >= R) return;

    float v[8];
    int cnt = 0;
    float m = -3.0e38f;
    for (int t = lane; t < D; t += 32) {
        float x = logits[(size_t)r * D + t];
        v[cnt++] = x;
        m = fmaxf(m, x);
    }
    #pragma unroll
    for (int off = 16; off; off >>= 1) m = fmaxf(m, __shfl_xor(m, off));
    float s = 0.0f;
    for (int i = 0; i < cnt; i++) { v[i] = __expf(v[i] - m); s += v[i]; }
    #pragma unroll
    for (int off = 16; off; off >>= 1) s += __shfl_xor(s, off);
    float inv = 1.0f / s;

    for (int q = 0; q < k; q++) {
        float best = -1.0f; int bi = 0;
        for (int i = 0; i < cnt; i++) if (v[i] > best) { best = v[i]; bi = i; }
        int bt = lane + bi * 32;
        #pragma unroll
        for (int off = 16; off; off >>= 1) {
            float ob = __shfl_xor(best, off);
            int obt  = __shfl_xor(bt, off);
            if (ob > best || (ob == best && obt < bt)) { best = ob; bt = obt; }
        }
        if ((bt & 31) == lane) v[bt >> 5] = -2.0f;   // remove selected
        if (lane == 0) { sc[(size_t)r * k + q] = best * inv; ix[(size_t)r * k + q] = bt; }
    }
}

// ------------------ final joint scores + stable sort -----------------------

__global__ void k_final(
    const float* s0, const int* x0, const float* s1, const int* x1,
    const float* s2, const int* x2, const float* s3, const int* x3,
    const float* s4, const int* x4, int* __restrict__ out)
{
    int b = blockIdx.x * blockDim.x + threadIdx.x;
    if (b >= 8192) return;
    float sc[15]; int ord[15];
    float a0 = s0[b];
    for (int c = 0; c < 15; c++)
        sc[c] = a0 * s1[b * 3 + c / 5] * s2[b * 3 + c / 5] * s3[b * 15 + c] * s4[b * 15 + c];
    for (int c = 0; c < 15; c++) ord[c] = c;
    for (int i = 1; i < 15; i++) {                 // stable insertion sort, descending
        int oc = ord[i]; float key = sc[oc];
        int p = i - 1;
        while (p >= 0 && sc[ord[p]] < key) { ord[p + 1] = ord[p]; p--; }
        ord[p + 1] = oc;
    }
    int* o = out + (size_t)b * 75;
    for (int rnk = 0; rnk < 15; rnk++) {
        int c = ord[rnk];
        o[rnk * 5 + 0] = x0[b];
        o[rnk * 5 + 1] = x1[b * 3 + c / 5];
        o[rnk * 5 + 2] = x2[b * 3 + c / 5];
        o[rnk * 5 + 3] = x3[b * 15 + c];
        o[rnk * 5 + 4] = x4[b * 15 + c];
    }
}

// ------------------------------- launcher ----------------------------------

extern "C" void kernel_launch(void* const* d_in, const int* in_sizes, int n_in,
                              void* d_out, int out_size, void* d_ws, size_t ws_size,
                              hipStream_t stream)
{
    (void)in_sizes; (void)n_in; (void)out_size; (void)ws_size;
    const int B = 8192, H = 1024;
    const int Kst[5]  = {1024, 1280, 1536, 1792, 2048};
    const int Nout[5] = {54, 87, 87, 235, 235};
    const int nst[5]  = {1, 1, 3, 3, 15};
    const int kst[5]  = {1, 3, 1, 5, 1};
    const int Rst[5]  = {8192, 8192, 24576, 24576, 122880};

    const float* features = (const float*)d_in[0];
    const float* Wmsg[4]  = {(const float*)d_in[1], (const float*)d_in[3],
                             (const float*)d_in[5], (const float*)d_in[7]};
    const float* bmsg[4]  = {(const float*)d_in[2], (const float*)d_in[4],
                             (const float*)d_in[6], (const float*)d_in[8]};
    const float *Wh1[5], *bh1[5], *Wh2[5], *bh2[5], *Wo[5], *bo[5];
    for (int i = 0; i < 5; i++) {
        Wh1[i] = (const float*)d_in[9  + 6 * i]; bh1[i] = (const float*)d_in[10 + 6 * i];
        Wh2[i] = (const float*)d_in[11 + 6 * i]; bh2[i] = (const float*)d_in[12 + 6 * i];
        Wo[i]  = (const float*)d_in[13 + 6 * i]; bo[i]  = (const float*)d_in[14 + 6 * i];
    }

    // ---- workspace layout
    char* ws = (char*)d_ws;
    size_t off = 0;
    auto alloc = [&](size_t bytes) -> void* {
        void* p = ws + off; off += (bytes + 255) & ~(size_t)255; return p;
    };
    ushort_t* featbf = (ushort_t*)alloc((size_t)B * H * 2);
    ushort_t *Wh1t[5], *Wh2t[5], *Wot[5];
    for (int i = 0; i < 5; i++) {
        Wh1t[i] = (ushort_t*)alloc((size_t)Kst[i] * H * 2);
        Wh2t[i] = (ushort_t*)alloc((size_t)H * H * 2);
        Wot[i]  = (ushort_t*)alloc((size_t)H * Nout[i] * 2);
    }
    ushort_t* X  = (ushort_t*)alloc((size_t)122880 * 2048 * 2);
    ushort_t* H1 = (ushort_t*)alloc((size_t)122880 * 1024 * 2);
    ushort_t* H2 = (ushort_t*)alloc((size_t)122880 * 1024 * 2);
    float* logits = (float*)alloc((size_t)122880 * 235 * 4);
    float* scp[5]; int* ixp[5];
    for (int i = 0; i < 5; i++) {
        scp[i] = (float*)alloc((size_t)B * nst[i] * kst[i] * 4);
        ixp[i] = (int*)  alloc((size_t)B * nst[i] * kst[i] * 4);
    }

    // ---- precision conversion (one pass; weights then live in L2 as bf16)
    k_cvt_bf16<<<(B * H + 255) / 256, 256, 0, stream>>>(features, featbf, B * H);
    for (int i = 0; i < 5; i++) {
        k_transpose_bf16<<<(Kst[i] * H + 255) / 256, 256, 0, stream>>>(Wh1[i], Wh1t[i], Kst[i], H);
        k_transpose_bf16<<<(H * H + 255) / 256, 256, 0, stream>>>(Wh2[i], Wh2t[i], H, H);
        k_transpose_bf16<<<(H * Nout[i] + 255) / 256, 256, 0, stream>>>(Wo[i], Wot[i], H, Nout[i]);
    }

    // ---- five beam stages
    for (int i = 0; i < 5; i++) {
        const ushort_t* Xi;
        if (i == 0) {
            Xi = featbf;
        } else {
            int dv[4] = {1, 1, 1, 1};
            const int wv[4] = {1, 3, 3, 15};
            if (i == 2)      { dv[0] = 3; }
            else if (i == 3) { dv[0] = 3; }
            else if (i == 4) { dv[0] = 15; dv[1] = 5; dv[2] = 5; }
            k_assemble<<<Rst[i], 256, 0, stream>>>(X, featbf, Kst[i], nst[i], i,
                ixp[0], dv[0], wv[0], Wmsg[0], bmsg[0],
                ixp[1], dv[1], wv[1], Wmsg[1], bmsg[1],
                ixp[2], dv[2], wv[2], Wmsg[2], bmsg[2],
                ixp[3], dv[3], wv[3], Wmsg[3], bmsg[3]);
            Xi = X;
        }
        dim3 g1(H / 128, Rst[i] / 128);
        k_gemm_bf16<0><<<g1, 256, 0, stream>>>(Xi, Wh1t[i], bh1[i], (void*)H1, Rst[i], H, Kst[i]);
        k_gemm_bf16<1><<<g1, 256, 0, stream>>>(H1, Wh2t[i], bh2[i], (void*)H2, Rst[i], H, 1024);
        dim3 g3((Nout[i] + 127) / 128, Rst[i] / 128);
        k_gemm_bf16<2><<<g3, 256, 0, stream>>>(H2, Wot[i], bo[i], (void*)logits, Rst[i], Nout[i], 1024);
        k_softmax_topk<<<(Rst[i] + 7) / 8, 256, 0, stream>>>(logits, Rst[i], Nout[i], kst[i], scp[i], ixp[i]);
    }

    k_final<<<(8192 + 255) / 256, 256, 0, stream>>>(
        scp[0], ixp[0], scp[1], ixp[1], scp[2], ixp[2],
        scp[3], ixp[3], scp[4], ixp[4], (int*)d_out);
}
